// IVDmodel_79053168050315
// MI455X (gfx1250) — matrix operand order repaired and checked by reading, not compiled, not run on hardware
//
#include <hip/hip_runtime.h>
#include <hip/hip_bf16.h>

// ---------------------------------------------------------------------------
// Model dims
// ---------------------------------------------------------------------------
#define BB   8
#define NN   512
#define LL   32
#define TT   32
#define FF   128
#define HH   128
#define EE   4096
#define NNF  5
#define H3   384

typedef __attribute__((ext_vector_type(16))) _Float16 v16h;
typedef __attribute__((ext_vector_type(8)))  _Float16 v8h;
typedef __attribute__((ext_vector_type(8)))  float    v8f;

// ---------------------------------------------------------------------------
// Workspace layout (bytes)
// ---------------------------------------------------------------------------
constexpr size_t A256(size_t x) { return (x + 255) & ~(size_t)255; }
constexpr size_t OFF_W_TREE_IOUX = 0;
constexpr size_t OFF_W_TREE_FX   = A256(OFF_W_TREE_IOUX + (size_t)H3*FF*2);
constexpr size_t OFF_W_TREE_IOUH = A256(OFF_W_TREE_FX   + (size_t)HH*FF*2);
constexpr size_t OFF_W_TREE_FH   = A256(OFF_W_TREE_IOUH + (size_t)H3*HH*2);
constexpr size_t OFF_W_GRU_IH    = A256(OFF_W_TREE_FH   + (size_t)HH*HH*2);
constexpr size_t OFF_W_GRU_HH    = A256(OFF_W_GRU_IH    + (size_t)4*H3*FF*2);
constexpr size_t OFF_W_COMB_IH   = A256(OFF_W_GRU_HH    + (size_t)4*H3*HH*2);
constexpr size_t OFF_W_COMB_HH   = A256(OFF_W_COMB_IH   + (size_t)2*H3*HH*2);
constexpr size_t OFF_W_CONNECT   = A256(OFF_W_COMB_HH   + (size_t)2*H3*HH*2);
constexpr size_t OFF_W_CONV01T   = A256(OFF_W_CONNECT   + (size_t)HH*2*NNF*HH*2);
constexpr size_t OFF_FEAT        = A256(OFF_W_CONV01T   + (size_t)2*HH*HH*2);
constexpr size_t OFF_COMB        = A256(OFF_FEAT        + (size_t)BB*NN*5*HH*2);
constexpr size_t OFF_VA          = A256(OFF_COMB        + (size_t)BB*NN*2*NNF*HH*2);
constexpr size_t OFF_VB          = A256(OFF_VA          + (size_t)BB*NN*HH*4);
constexpr size_t OFF_XW          = A256(OFF_VB          + (size_t)BB*NN*HH*4);
constexpr size_t OFF_DEG         = A256(OFF_XW          + (size_t)BB*NN*HH*4);

// ---------------------------------------------------------------------------
// Helpers
// ---------------------------------------------------------------------------
__device__ __forceinline__ v8f wmma16(v16h a, v16h b, v8f c) {
  // D = A(16x32 f16) x B(32x16 f16) + C(16x16 f32)
  return __builtin_amdgcn_wmma_f32_16x16x32_f16(
      false, a, false, b, (short)0, c, false, false);
}

__device__ __forceinline__ float sigm(float x) { return 1.0f / (1.0f + __expf(-x)); }

// A-fragment: per-lane pointer at (row base + k0 + 8*(lane>>4)) — two 8-elem K chunks.
__device__ __forceinline__ v16h frag_a_f32(const float* p) {
  v16h a;
#pragma unroll
  for (int i = 0; i < 8; ++i) { a[i] = (_Float16)p[i]; a[i + 8] = (_Float16)p[16 + i]; }
  return a;
}
__device__ __forceinline__ v16h frag_a_f16(const _Float16* p) {
  v16h a;
#pragma unroll
  for (int i = 0; i < 8; ++i) { a[i] = p[i]; a[i + 8] = p[16 + i]; }
  return a;
}
// B-fragment: per-lane pointer at (k0 + 8*(lane>>4)) — two 8-elem K chunks.
__device__ __forceinline__ v16h frag_b(const _Float16* p) {
  v8h lo = *(const v8h*)p;
  v8h hi = *(const v8h*)(p + 16);
  return __builtin_shufflevector(lo, hi, 0, 1, 2, 3, 4, 5, 6, 7, 8, 9, 10, 11, 12, 13, 14, 15);
}

// ---------------------------------------------------------------------------
// Weight conversion
// ---------------------------------------------------------------------------
__global__ void k_cvt(const float* __restrict__ s, _Float16* __restrict__ d, int n) {
  int i = blockIdx.x * blockDim.x + threadIdx.x;
  if (i < n) d[i] = (_Float16)s[i];
}
// conv_W01 is [2][K][Nout] for x@W; store transposed [2][Nout][K] for B-fragments.
__global__ void k_cvtT(const float* __restrict__ s, _Float16* __restrict__ d) {
  int i = blockIdx.x * blockDim.x + threadIdx.x;
  if (i >= 2 * HH * HH) return;
  int l = i / (HH * HH), r = i % (HH * HH);
  int k = r / HH, n = r % HH;
  d[l * HH * HH + n * HH + k] = (_Float16)s[i];
}

__global__ void k_fill(float* __restrict__ p, float v, int n) {
  int i = blockIdx.x * blockDim.x + threadIdx.x;
  if (i < n) p[i] = v;
}

// ---------------------------------------------------------------------------
// TreeLSTM chain scan: one block per (batch, 128-row tile); 16 waves.
// Wave w: row-tile mt = w&7, hidden-col half nh = w>>3 (4 nt tiles each).
// Gate accumulators start at 0 (WMMA inline C=0); biases are added in the
// scalar epilogue so nothing bias-related is hoisted/spilled across t.
// `wstep` is 0 at runtime; makes weight pointers loop-variant to defeat LICM.
// ---------------------------------------------------------------------------
__global__ __launch_bounds__(512) void k_tree(
    const float* __restrict__ x,        // [B][N][L][F]
    const _Float16* __restrict__ wioux, const _Float16* __restrict__ wfx,
    const _Float16* __restrict__ wiouh, const _Float16* __restrict__ wfh,
    const float* __restrict__ bioux, const float* __restrict__ biouh,
    const float* __restrict__ bfx,   const float* __restrict__ bfh,
    _Float16* __restrict__ feat,        // [B][N][5][H]
    long wstep)
{
  __shared__ _Float16 hbuf[2][HH * HH];   // 64 KB double-buffered hidden state
  const int tile = blockIdx.x & 3;
  const int b    = blockIdx.x >> 2;
  const int row0 = tile * 128;
  const int lane = threadIdx.x & 31;
  const int wv   = threadIdx.x >> 5;      // 0..15
  const int mt   = wv & 7;                // row tile
  const int nh   = wv >> 3;               // hidden-col half (0..1)

  for (int i = threadIdx.x; i < HH * HH; i += 512) hbuf[0][i] = (_Float16)0.0f;
  __syncthreads();

  v8f c[4];
#pragma unroll
  for (int q = 0; q < 4; ++q)
#pragma unroll
    for (int v = 0; v < 8; ++v) c[q][v] = 0.0f;

  const int arow  = mt * 16 + (lane & 15);
  const int akoff = (lane >> 4) << 3;   // 0 or 8 (A chunk)
  const int bcol  = lane & 15;
  const int bkoff = (lane >> 4) << 3;   // 0 or 8 (B chunk)
  const float* xrow = x + (((size_t)b * NN + row0 + arow) * LL) * FF;

  // Loop-variant (opaque) weight pointers: wstep == 0 at runtime.
  const _Float16* pwx  = wioux;
  const _Float16* pwfx = wfx;
  const _Float16* pwh  = wiouh;
  const _Float16* pwfh = wfh;

  for (int t = 0; t < LL; ++t) {
    const _Float16* hprev = hbuf[t & 1];
    _Float16*       hnext = hbuf[(t + 1) & 1];

    v16h ax[4], ah[4];
#pragma unroll
    for (int kk = 0; kk < 4; ++kk) {
      ax[kk] = frag_a_f32(xrow + (size_t)t * FF + kk * 32 + akoff);
      ah[kk] = frag_a_f16(hprev + arow * HH + kk * 32 + akoff);
    }

#pragma unroll
    for (int q = 0; q < 4; ++q) {
      const int nt  = nh * 4 + q;
      const int col = nt * 16 + bcol;
      const float bi = bioux[col]       + biouh[col];
      const float bo = bioux[128 + col] + biouh[128 + col];
      const float bu = bioux[256 + col] + biouh[256 + col];
      const float bf = bfx[col]         + bfh[col];
      v8f ai = {}, ao = {}, au = {}, af = {};

#pragma unroll
      for (int kk = 0; kk < 4; ++kk) {
        const int k0 = kk * 32 + bkoff;
        ai = wmma16(ax[kk], frag_b(pwx  + (size_t)(col)       * FF + k0), ai);
        ao = wmma16(ax[kk], frag_b(pwx  + (size_t)(col + 128) * FF + k0), ao);
        au = wmma16(ax[kk], frag_b(pwx  + (size_t)(col + 256) * FF + k0), au);
        af = wmma16(ax[kk], frag_b(pwfx + (size_t)(col)       * FF + k0), af);
        ai = wmma16(ah[kk], frag_b(pwh  + (size_t)(col)       * HH + k0), ai);
        ao = wmma16(ah[kk], frag_b(pwh  + (size_t)(col + 128) * HH + k0), ao);
        au = wmma16(ah[kk], frag_b(pwh  + (size_t)(col + 256) * HH + k0), au);
        af = wmma16(ah[kk], frag_b(pwfh + (size_t)(col)       * HH + k0), af);
      }

#pragma unroll
      for (int v = 0; v < 8; ++v) {
        const float iv = sigm(ai[v] + bi);
        const float ov = sigm(ao[v] + bo);
        const float uv = tanhf(au[v] + bu);
        const float fv = sigm(af[v] + bf);
        const float cv = iv * uv + fv * c[q][v];
        c[q][v] = cv;
        const float hv = ov * tanhf(cv);
        hnext[(mt * 16 + v + ((lane >> 4) << 3)) * HH + col] = (_Float16)hv;
      }
    }
    __syncthreads();
    pwx += wstep; pwfx += wstep; pwh += wstep; pwfh += wstep;
  }

  const _Float16* hf = hbuf[0];  // buffer written at t = L-1
  for (int i = threadIdx.x; i < HH * HH; i += 512) {
    int r = i >> 7, cc = i & 127;
    feat[(((size_t)b * NN + row0 + r) * 5 + 0) * HH + cc] = hf[i];
  }
}

// ---------------------------------------------------------------------------
// 4 unidirectional GRUs: grid = B*4*(N/128); 16 waves per block.
// ---------------------------------------------------------------------------
__global__ __launch_bounds__(512) void k_gru(
    const float* __restrict__ x,        // [B][4][N][T][F]
    const _Float16* __restrict__ wih,   // [4][384][128]
    const _Float16* __restrict__ whh,
    const float* __restrict__ bih, const float* __restrict__ bhh,  // [4][384]
    _Float16* __restrict__ feat,
    long wstep)
{
  __shared__ _Float16 hbuf[2][HH * HH];
  const int tile = blockIdx.x & 3;
  const int g    = (blockIdx.x >> 2) & 3;
  const int b    = blockIdx.x >> 4;
  const int row0 = tile * 128;
  const int lane = threadIdx.x & 31;
  const int wv   = threadIdx.x >> 5;
  const int mt   = wv & 7;
  const int nh   = wv >> 3;

  for (int i = threadIdx.x; i < HH * HH; i += 512) hbuf[0][i] = (_Float16)0.0f;
  __syncthreads();

  const _Float16* Wih = wih + (size_t)g * H3 * FF;
  const _Float16* Whh = whh + (size_t)g * H3 * HH;
  const float*    Bih = bih + g * H3;
  const float*    Bhh = bhh + g * H3;

  const int arow  = mt * 16 + (lane & 15);
  const int akoff = (lane >> 4) << 3;
  const int bcol  = lane & 15;
  const int bkoff = (lane >> 4) << 3;
  const float* xrow = x + ((((size_t)b * 4 + g) * NN + row0 + arow) * TT) * FF;

  for (int t = 0; t < TT; ++t) {
    const _Float16* hprev = hbuf[t & 1];
    _Float16*       hnext = hbuf[(t + 1) & 1];

    v16h ax[4], ah[4];
#pragma unroll
    for (int kk = 0; kk < 4; ++kk) {
      ax[kk] = frag_a_f32(xrow + (size_t)t * FF + kk * 32 + akoff);
      ah[kk] = frag_a_f16(hprev + arow * HH + kk * 32 + akoff);
    }

#pragma unroll
    for (int q = 0; q < 4; ++q) {
      const int nt  = nh * 4 + q;
      const int col = nt * 16 + bcol;
      const float br = Bih[col]       + Bhh[col];
      const float bz = Bih[128 + col] + Bhh[128 + col];
      const float bn = Bih[256 + col];
      const float bh = Bhh[256 + col];
      v8f ar = {}, az = {}, an = {}, hn = {};

#pragma unroll
      for (int kk = 0; kk < 4; ++kk) {
        const int k0 = kk * 32 + bkoff;
        ar = wmma16(ax[kk], frag_b(Wih + (size_t)(col)       * FF + k0), ar);
        az = wmma16(ax[kk], frag_b(Wih + (size_t)(col + 128) * FF + k0), az);
        an = wmma16(ax[kk], frag_b(Wih + (size_t)(col + 256) * FF + k0), an);
        ar = wmma16(ah[kk], frag_b(Whh + (size_t)(col)       * HH + k0), ar);
        az = wmma16(ah[kk], frag_b(Whh + (size_t)(col + 128) * HH + k0), az);
        hn = wmma16(ah[kk], frag_b(Whh + (size_t)(col + 256) * HH + k0), hn);
      }

#pragma unroll
      for (int v = 0; v < 8; ++v) {
        const int rrow = mt * 16 + v + ((lane >> 4) << 3);
        const float hp = (float)hprev[rrow * HH + col];
        const float r  = sigm(ar[v] + br);
        const float z  = sigm(az[v] + bz);
        const float nn = tanhf(an[v] + bn + r * (hn[v] + bh));
        hnext[rrow * HH + col] = (_Float16)((1.0f - z) * nn + z * hp);
      }
    }
    __syncthreads();
    Wih += wstep; Whh += wstep;
  }

  const _Float16* hf = hbuf[0];
  for (int i = threadIdx.x; i < HH * HH; i += 512) {
    int r = i >> 7, cc = i & 127;
    feat[(((size_t)b * NN + row0 + r) * 5 + 1 + g) * HH + cc] = hf[i];
  }
}

// ---------------------------------------------------------------------------
// Bidirectional combine GRU over 5 steps: grid = B*2*(N/128); 16 waves.
// ---------------------------------------------------------------------------
__global__ __launch_bounds__(512) void k_comb(
    const _Float16* __restrict__ feat,  // [B][N][5][H]
    const _Float16* __restrict__ wih,   // [2][384][128]
    const _Float16* __restrict__ whh,
    const float* __restrict__ bih, const float* __restrict__ bhh,  // [2][384]
    _Float16* __restrict__ comb,        // [B][N][1280]
    long wstep)
{
  __shared__ _Float16 hbuf[2][HH * HH];
  const int tile = blockIdx.x & 3;
  const int dir  = (blockIdx.x >> 2) & 1;
  const int b    = blockIdx.x >> 3;
  const int row0 = tile * 128;
  const int lane = threadIdx.x & 31;
  const int wv   = threadIdx.x >> 5;
  const int mt   = wv & 7;
  const int nh   = wv >> 3;

  for (int i = threadIdx.x; i < HH * HH; i += 512) hbuf[0][i] = (_Float16)0.0f;
  __syncthreads();

  const _Float16* Wih = wih + (size_t)dir * H3 * HH;
  const _Float16* Whh = whh + (size_t)dir * H3 * HH;
  const float*    Bih = bih + dir * H3;
  const float*    Bhh = bhh + dir * H3;

  const int arow  = mt * 16 + (lane & 15);
  const int akoff = (lane >> 4) << 3;
  const int bcol  = lane & 15;
  const int bkoff = (lane >> 4) << 3;

  for (int s = 0; s < 5; ++s) {
    const int tin = dir ? (4 - s) : s;
    const _Float16* hprev = hbuf[s & 1];
    _Float16*       hnext = hbuf[(s + 1) & 1];
    const _Float16* xp = feat + (((size_t)b * NN + row0 + arow) * 5 + tin) * HH;

    v16h ax[4], ah[4];
#pragma unroll
    for (int kk = 0; kk < 4; ++kk) {
      ax[kk] = frag_a_f16(xp + kk * 32 + akoff);
      ah[kk] = frag_a_f16(hprev + arow * HH + kk * 32 + akoff);
    }

#pragma unroll
    for (int q = 0; q < 4; ++q) {
      const int nt  = nh * 4 + q;
      const int col = nt * 16 + bcol;
      const float br = Bih[col]       + Bhh[col];
      const float bz = Bih[128 + col] + Bhh[128 + col];
      const float bn = Bih[256 + col];
      const float bh = Bhh[256 + col];
      v8f ar = {}, az = {}, an = {}, hn = {};

#pragma unroll
      for (int kk = 0; kk < 4; ++kk) {
        const int k0 = kk * 32 + bkoff;
        ar = wmma16(ax[kk], frag_b(Wih + (size_t)(col)       * HH + k0), ar);
        az = wmma16(ax[kk], frag_b(Wih + (size_t)(col + 128) * HH + k0), az);
        an = wmma16(ax[kk], frag_b(Wih + (size_t)(col + 256) * HH + k0), an);
        ar = wmma16(ah[kk], frag_b(Whh + (size_t)(col)       * HH + k0), ar);
        az = wmma16(ah[kk], frag_b(Whh + (size_t)(col + 128) * HH + k0), az);
        hn = wmma16(ah[kk], frag_b(Whh + (size_t)(col + 256) * HH + k0), hn);
      }

#pragma unroll
      for (int v = 0; v < 8; ++v) {
        const int rrow = mt * 16 + v + ((lane >> 4) << 3);
        const float hp = (float)hprev[rrow * HH + col];
        const float r  = sigm(ar[v] + br);
        const float z  = sigm(az[v] + bz);
        const float nn = tanhf(an[v] + bn + r * (hn[v] + bh));
        const float hv = (1.0f - z) * nn + z * hp;
        hnext[rrow * HH + col] = (_Float16)hv;
        comb[((size_t)b * NN + row0 + rrow) * (2 * NNF * HH) + tin * 256 + dir * 128 + col] =
            (_Float16)hv;
      }
    }
    __syncthreads();
    Wih += wstep; Whh += wstep;
  }
}

// ---------------------------------------------------------------------------
// connect: [N,1280] x [1280,128] (+bias) -> vA fp32.  K-outer, nt-inner with
// acc[8] so no large invariant load sets exist to hoist/spill.
// ---------------------------------------------------------------------------
__global__ __launch_bounds__(256) void k_connect(
    const _Float16* __restrict__ comb, const _Float16* __restrict__ w,  // [128][1280]
    const float* __restrict__ bias, float* __restrict__ out)            // [B][N][128]
{
  const int tile = blockIdx.x & 3;
  const int b    = blockIdx.x >> 2;
  const int row0 = tile * 128;
  const int lane = threadIdx.x & 31;
  const int mt   = threadIdx.x >> 5;
  const int arow  = mt * 16 + (lane & 15);
  const int akoff = (lane >> 4) << 3;
  const int bcol  = lane & 15;
  const int bkoff = (lane >> 4) << 3;
  const _Float16* ap = comb + ((size_t)b * NN + row0 + arow) * (2 * NNF * HH);

  v8f acc[8];
#pragma unroll
  for (int nt = 0; nt < 8; ++nt)
#pragma unroll
    for (int v = 0; v < 8; ++v) acc[nt][v] = 0.0f;

  for (int kk = 0; kk < 40; ++kk) {
    const int k0 = kk * 32;
    v16h a = frag_a_f16(ap + k0 + akoff);
#pragma unroll
    for (int nt = 0; nt < 8; ++nt) {
      const int col = nt * 16 + bcol;
      v16h bb = frag_b(w + (size_t)col * (2 * NNF * HH) + k0 + bkoff);
      acc[nt] = wmma16(a, bb, acc[nt]);
    }
  }

#pragma unroll
  for (int nt = 0; nt < 8; ++nt) {
    const int col = nt * 16 + bcol;
    const float bv = bias[col];
#pragma unroll
    for (int v = 0; v < 8; ++v)
      out[((size_t)b * NN + row0 + mt * 16 + v + ((lane >> 4) << 3)) * HH + col] =
          acc[nt][v] + bv;
  }
}

// ---------------------------------------------------------------------------
// GCN x@W GEMM (W pre-transposed to [out][in] f16): [N,128]x[128,128]
// ---------------------------------------------------------------------------
__global__ __launch_bounds__(256) void k_xw(
    const float* __restrict__ vin, const _Float16* __restrict__ wT,
    float* __restrict__ xw)
{
  const int tile = blockIdx.x & 3;
  const int b    = blockIdx.x >> 2;
  const int row0 = tile * 128;
  const int lane = threadIdx.x & 31;
  const int mt   = threadIdx.x >> 5;
  const int arow  = mt * 16 + (lane & 15);
  const int akoff = (lane >> 4) << 3;
  const int bcol  = lane & 15;
  const int bkoff = (lane >> 4) << 3;
  const float* ap = vin + ((size_t)b * NN + row0 + arow) * HH;

  v8f acc[8];
#pragma unroll
  for (int nt = 0; nt < 8; ++nt)
#pragma unroll
    for (int v = 0; v < 8; ++v) acc[nt][v] = 0.0f;

#pragma unroll
  for (int kk = 0; kk < 4; ++kk) {
    v16h a = frag_a_f32(ap + kk * 32 + akoff);
#pragma unroll
    for (int nt = 0; nt < 8; ++nt) {
      const int col = nt * 16 + bcol;
      v16h bb = frag_b(wT + (size_t)col * HH + kk * 32 + bkoff);
      acc[nt] = wmma16(a, bb, acc[nt]);
    }
  }

#pragma unroll
  for (int nt = 0; nt < 8; ++nt) {
    const int col = nt * 16 + bcol;
#pragma unroll
    for (int v = 0; v < 8; ++v)
      xw[((size_t)b * NN + row0 + mt * 16 + v + ((lane >> 4) << 3)) * HH + col] = acc[nt][v];
  }
}

// ---------------------------------------------------------------------------
// GCN degree (deg pre-filled with 1.0 for self loops) and aggregation
// ---------------------------------------------------------------------------
__global__ void k_deg(const int* __restrict__ ei, float* __restrict__ deg) {
  int i = blockIdx.x * blockDim.x + threadIdx.x;
  if (i >= BB * EE) return;
  int b = i / EE, e = i % EE;
  int d = ei[((size_t)b * 2 + 1) * EE + e];
  atomicAdd(&deg[b * NN + d], 1.0f);
}

__global__ void k_agg(const int* __restrict__ ei, const float* __restrict__ deg,
                      const float* __restrict__ xw, float* __restrict__ out) {
  int idx = blockIdx.x * blockDim.x + threadIdx.x;   // B*(E+N)*32 threads, 4 feats each
  int fg = idx & 31;
  int rest = idx >> 5;
  int e = rest % (EE + NN);
  int b = rest / (EE + NN);
  if (b >= BB) return;
  int s, d;
  if (e < EE) {
    s = ei[((size_t)b * 2) * EE + e];
    d = ei[((size_t)b * 2 + 1) * EE + e];
  } else {
    s = d = e - EE;   // self loop
  }
  float nrm = rsqrtf(deg[b * NN + s]) * rsqrtf(deg[b * NN + d]);
  const float* xs = xw + ((size_t)b * NN + s) * HH + fg * 4;
  float* od = out + ((size_t)b * NN + d) * HH + fg * 4;
#pragma unroll
  for (int i = 0; i < 4; ++i) atomicAdd(od + i, nrm * xs[i]);
}

__global__ void k_bias_relu(float* __restrict__ p, const float* __restrict__ bias) {
  int i = blockIdx.x * blockDim.x + threadIdx.x;
  if (i >= BB * NN * HH) return;
  float v = p[i] + bias[i & 127];
  p[i] = fmaxf(v, 0.0f);
}

// ---------------------------------------------------------------------------
// Final: out5 = v @ conv_W2 + b2; global max over nodes; softmax -> d_out
// ---------------------------------------------------------------------------
__global__ __launch_bounds__(256) void k_final(
    const float* __restrict__ v, const float* __restrict__ w2,
    const float* __restrict__ b2, float* __restrict__ out)
{
  __shared__ float red[256 * NNF];
  const int b = blockIdx.x;
  float m[NNF];
#pragma unroll
  for (int j = 0; j < NNF; ++j) m[j] = -3.4e38f;

  for (int n = threadIdx.x; n < NN; n += 256) {
    const float* vp = v + ((size_t)b * NN + n) * HH;
    float o[NNF];
#pragma unroll
    for (int j = 0; j < NNF; ++j) o[j] = b2[j];
    for (int k = 0; k < HH; ++k) {
      float xv = vp[k];
#pragma unroll
      for (int j = 0; j < NNF; ++j) o[j] += xv * w2[k * NNF + j];
    }
#pragma unroll
    for (int j = 0; j < NNF; ++j) m[j] = fmaxf(m[j], o[j]);
  }
#pragma unroll
  for (int j = 0; j < NNF; ++j) red[threadIdx.x * NNF + j] = m[j];
  __syncthreads();
  for (int s = 128; s > 0; s >>= 1) {
    if ((int)threadIdx.x < s)
#pragma unroll
      for (int j = 0; j < NNF; ++j)
        red[threadIdx.x * NNF + j] =
            fmaxf(red[threadIdx.x * NNF + j], red[(threadIdx.x + s) * NNF + j]);
    __syncthreads();
  }
  if (threadIdx.x == 0) {
    float mx = red[0];
#pragma unroll
    for (int j = 1; j < NNF; ++j) mx = fmaxf(mx, red[j]);
    float e[NNF], sum = 0.0f;
#pragma unroll
    for (int j = 0; j < NNF; ++j) { e[j] = __expf(red[j] - mx); sum += e[j]; }
#pragma unroll
    for (int j = 0; j < NNF; ++j) out[b * NNF + j] = e[j] / sum;
  }
}

// ---------------------------------------------------------------------------
// Host launcher
// ---------------------------------------------------------------------------
extern "C" void kernel_launch(void* const* d_in, const int* in_sizes, int n_in,
                              void* d_out, int out_size, void* d_ws, size_t ws_size,
                              hipStream_t stream) {
  (void)in_sizes; (void)n_in; (void)out_size; (void)ws_size;
  char* ws = (char*)d_ws;
  auto F16 = [&](size_t off) { return (_Float16*)(ws + off); };
  auto F32 = [&](size_t off) { return (float*)(ws + off); };

  const float* tree_x = (const float*)d_in[0];
  const float* seq_x  = (const float*)d_in[1];
  const int*   ei     = (const int*)d_in[2];

  auto cvt = [&](int src_idx, size_t off, int n) {
    k_cvt<<<(n + 255) / 256, 256, 0, stream>>>((const float*)d_in[src_idx], F16(off), n);
  };
  cvt(3,  OFF_W_TREE_IOUX, H3 * FF);
  cvt(7,  OFF_W_TREE_FX,   HH * FF);
  cvt(5,  OFF_W_TREE_IOUH, H3 * HH);
  cvt(9,  OFF_W_TREE_FH,   HH * HH);
  cvt(11, OFF_W_GRU_IH,    4 * H3 * FF);
  cvt(12, OFF_W_GRU_HH,    4 * H3 * HH);
  cvt(15, OFF_W_COMB_IH,   2 * H3 * HH);
  cvt(16, OFF_W_COMB_HH,   2 * H3 * HH);
  cvt(19, OFF_W_CONNECT,   HH * 2 * NNF * HH);
  k_cvtT<<<(2 * HH * HH + 255) / 256, 256, 0, stream>>>((const float*)d_in[21],
                                                        F16(OFF_W_CONV01T));

  const long wstep = 0;   // opaque to device code; defeats LICM of weight loads

  // Scans (WMMA) — 512 threads = 16 waves per block
  k_tree<<<BB * 4, 512, 0, stream>>>(
      tree_x, F16(OFF_W_TREE_IOUX), F16(OFF_W_TREE_FX), F16(OFF_W_TREE_IOUH),
      F16(OFF_W_TREE_FH), (const float*)d_in[4], (const float*)d_in[6],
      (const float*)d_in[8], (const float*)d_in[10], F16(OFF_FEAT), wstep);

  k_gru<<<BB * 4 * 4, 512, 0, stream>>>(
      seq_x, F16(OFF_W_GRU_IH), F16(OFF_W_GRU_HH),
      (const float*)d_in[13], (const float*)d_in[14], F16(OFF_FEAT), wstep);

  k_comb<<<BB * 2 * 4, 512, 0, stream>>>(
      F16(OFF_FEAT), F16(OFF_W_COMB_IH), F16(OFF_W_COMB_HH),
      (const float*)d_in[17], (const float*)d_in[18], F16(OFF_COMB), wstep);

  k_connect<<<BB * 4, 256, 0, stream>>>(
      F16(OFF_COMB), F16(OFF_W_CONNECT), (const float*)d_in[20], F32(OFF_VA));

  // GCN degree (self loops contribute 1)
  k_fill<<<(BB * NN + 255) / 256, 256, 0, stream>>>(F32(OFF_DEG), 1.0f, BB * NN);
  k_deg<<<(BB * EE + 255) / 256, 256, 0, stream>>>(ei, F32(OFF_DEG));

  const int nAgg = BB * (EE + NN) * 32;
  const int nBR  = BB * NN * HH;

  // Layer 1: vA -> vB
  k_xw<<<BB * 4, 256, 0, stream>>>(F32(OFF_VA), F16(OFF_W_CONV01T), F32(OFF_XW));
  k_fill<<<(nBR + 255) / 256, 256, 0, stream>>>(F32(OFF_VB), 0.0f, nBR);
  k_agg<<<(nAgg + 255) / 256, 256, 0, stream>>>(ei, F32(OFF_DEG), F32(OFF_XW), F32(OFF_VB));
  k_bias_relu<<<(nBR + 255) / 256, 256, 0, stream>>>(F32(OFF_VB), (const float*)d_in[22]);

  // Layer 2: vB -> vA
  k_xw<<<BB * 4, 256, 0, stream>>>(F32(OFF_VB), F16(OFF_W_CONV01T) + HH * HH, F32(OFF_XW));
  k_fill<<<(nBR + 255) / 256, 256, 0, stream>>>(F32(OFF_VA), 0.0f, nBR);
  k_agg<<<(nAgg + 255) / 256, 256, 0, stream>>>(ei, F32(OFF_DEG), F32(OFF_XW), F32(OFF_VA));
  k_bias_relu<<<(nBR + 255) / 256, 256, 0, stream>>>(F32(OFF_VA), (const float*)d_in[22] + HH);

  // Final projection + max pool + softmax
  k_final<<<BB, 256, 0, stream>>>(F32(OFF_VA), (const float*)d_in[23],
                                  (const float*)d_in[24], (float*)d_out);
}
